// BeamDecoder_24618752540961
// MI455X (gfx1250) — matrix operand order repaired and checked by reading, not compiled, not run on hardware
//
#include <hip/hip_runtime.h>

// ---------------------------------------------------------------------------
// Beam decoder for MI455X (gfx1250, wave32, WMMA).
// Per step: logits = (token_emb[last] + enc_pool) @ W_out  [64 x 1024 x 32000]
// W_out (131MB f32) is L2-resident (192MB L2); each step reads it exactly once.
// GEMM runs as bf16 hi/lo split (3x v_wmma_f32_16x16x32_bf16) for ~f32 logits.
// Workspace use: ~8.6 MB.
// ---------------------------------------------------------------------------

typedef __attribute__((ext_vector_type(16))) __bf16 v16bf;
typedef __attribute__((ext_vector_type(8)))  float  v8f;

#define NB     16      // batch
#define BEAM   4
#define DD     1024    // model dim
#define VV     32000   // vocab
#define TSTEPS 32
#define MM     64      // NB*BEAM rows
#define NEGBIG -1e9f
#define FLOW   -3.0e38f

// workspace offsets (bytes)
#define OFF_POOL    0u                       // 16*1024*4      = 65536
#define OFF_AHI     65536u                   // 64*1024*2      = 131072
#define OFF_ALO     196608u                  // 131072
#define OFF_LOGITS  327680u                  // 64*32000*4     = 8192000
#define OFF_PART    8519680u                 // 64*16*40       = 40960
#define OFF_SCORES  8560640u                 // 64*4
#define OFF_TOKENS  8560896u                 // 16*4*32*4      = 8192
#define OFF_LAST    8569088u                 // 64*4

struct RowChunk { float mx; float sm; float v[4]; int ix[4]; };

__device__ __forceinline__ bool better(float v1, int i1, float v2, int i2) {
  return (v1 > v2) || (v1 == v2 && i1 < i2);
}

__device__ __forceinline__ void ins4(float* v, int* ix, float x, int id) {
  if (!better(x, id, v[3], ix[3])) return;
  v[3] = x; ix[3] = id;
  #pragma unroll
  for (int q = 3; q > 0; --q) {
    if (better(v[q], ix[q], v[q-1], ix[q-1])) {
      float tv = v[q]; v[q] = v[q-1]; v[q-1] = tv;
      int   ti = ix[q]; ix[q] = ix[q-1]; ix[q-1] = ti;
    }
  }
}

// merge sorted-desc 4-list b into a (both sorted), result sorted-desc top4 in a
__device__ __forceinline__ void merge4(float* av, int* ai,
                                       const float* bv, const int* bi) {
  float ov[4]; int oi[4];
  int pa = 0, pb = 0;
  #pragma unroll
  for (int q = 0; q < 4; ++q) {
    if (better(av[pa], ai[pa], bv[pb], bi[pb])) { ov[q] = av[pa]; oi[q] = ai[pa]; ++pa; }
    else                                        { ov[q] = bv[pb]; oi[q] = bi[pb]; ++pb; }
  }
  #pragma unroll
  for (int q = 0; q < 4; ++q) { av[q] = ov[q]; ai[q] = oi[q]; }
}

// Write one A element (row,k) into the WMMA 16-bit A-fragment layout:
// lane l<16: row=l, elems j: K=(j&7)+16*(j>>3); lanes>=16: +8 on K.
__device__ __forceinline__ void writeAfrag(__bf16* Ahi, __bf16* Alo,
                                           int row, int k, float val) {
  int mt = row >> 4, rl = row & 15;
  int kt = k >> 5,  kk = k & 31;
  int h  = (kk >> 3) & 1;
  int j  = (kk & 7) | (((kk >> 4) & 1) << 3);
  int lane = rl + (h << 4);
  int idx  = ((mt * 32 + kt) * 32 + lane) * 16 + j;
  __bf16 hi = (__bf16)val;
  float  r  = val - (float)hi;
  Ahi[idx] = hi;
  Alo[idx] = (__bf16)r;
}

__device__ __forceinline__ void buildA(const float* __restrict__ emb,
                                       const float* __restrict__ pool,
                                       const int* __restrict__ last,
                                       __bf16* Ahi, __bf16* Alo, int t) {
  for (int idx = t; idx < MM * DD; idx += 256) {
    int row = idx >> 10;
    int k   = idx & 1023;
    float v = emb[(size_t)last[row] * DD + k] + pool[(row >> 2) * DD + k];
    writeAfrag(Ahi, Alo, row, k, v);
  }
}

// ---------------------------------------------------------------------------
__global__ void k_encpool(const float* __restrict__ enc,
                          const float* __restrict__ mask,
                          float* __restrict__ pool) {
  int b = blockIdx.x, t = threadIdx.x;
  float msum = 0.f;
  for (int s = 0; s < 128; ++s) msum += mask[b * 128 + s];
  float inv = 1.f / (msum + 1e-6f);
  for (int d = t; d < DD; d += 256) {
    float acc = 0.f;
    for (int s = 0; s < 128; ++s)
      acc += enc[((size_t)b * 128 + s) * DD + d] * mask[b * 128 + s];
    pool[b * DD + d] = acc * inv;
  }
}

__global__ void k_init(const int* __restrict__ tgt,
                       const float* __restrict__ emb,
                       const float* __restrict__ pool,
                       int* tokens, float* scores, int* last,
                       __bf16* Ahi, __bf16* Alo) {
  int t = threadIdx.x;
  for (int i = t; i < NB * BEAM * TSTEPS; i += 256) tokens[i] = 0;
  __syncthreads();
  if (t < MM) {
    int b = t >> 2, k = t & 3;
    tokens[(b * BEAM + k) * TSTEPS + 0] = tgt[b];
    scores[t] = (k == 0) ? 0.f : NEGBIG;
    last[t]   = tgt[b];
  }
  __syncthreads();
  buildA(emb, pool, last, Ahi, Alo, t);
}

// ---------------------------------------------------------------------------
// GEMM: 125 blocks x 256 threads. Block covers all 64 rows x 256 vocab cols.
// A frags (hi/lo) staged through LDS per k-tile; W converted f32->bf16 hi/lo
// in registers; 6 WMMAs per (m-tile, n-subtile... ) per k-tile.
// ---------------------------------------------------------------------------
__global__ void __launch_bounds__(256)
k_gemm(const float* __restrict__ W,
       const __bf16* __restrict__ Ahi, const __bf16* __restrict__ Alo,
       float* __restrict__ logits) {
  __shared__ __bf16 sA[8][32][16] __attribute__((aligned(32))); // [m*2+hl][lane][16]
  int t    = threadIdx.x;
  int lane = t & 31;
  int w    = t >> 5;                       // wave 0..7
  int v0   = blockIdx.x * 256 + w * 32;    // this wave's 32-col strip
  int nlo  = lane & 15;
  int hh   = lane >> 4;

  v8f acc[4][2];
  v8f zero = {};
  #pragma unroll
  for (int m = 0; m < 4; ++m)
    #pragma unroll
    for (int s = 0; s < 2; ++s) acc[m][s] = zero;

  for (int kt = 0; kt < 32; ++kt) {
    __syncthreads();
    {
      int f  = t >> 5;                 // 0..7 : frag (m = f>>1, hilo = f&1)
      int ln = t & 31;
      const __bf16* src = (f & 1) ? Alo : Ahi;
      int mt = f >> 1;
      const uint4* s4 = (const uint4*)(src + ((size_t)(mt * 32 + kt) * 32 + ln) * 16);
      uint4* d4 = (uint4*)(&sA[f][ln][0]);
      d4[0] = s4[0];
      d4[1] = s4[1];
    }
    __syncthreads();

    // Build B fragments (32x16 bf16 each) for 2 n-subtiles, hi and lo parts.
    // B layout: lane holds col = nlo, elems j: K = (j&7) + 8*(lane>>4) + 16*(j>>3).
    v16bf bhi[2], blo[2];
    #pragma unroll
    for (int s = 0; s < 2; ++s) {
      int col = v0 + s * 16 + nlo;
      #pragma unroll
      for (int j = 0; j < 16; ++j) {
        int K = kt * 32 + (j & 7) + 8 * hh + 16 * (j >> 3);
        float x = W[(size_t)K * VV + col];
        __bf16 hi = (__bf16)x;
        float  r  = x - (float)hi;
        bhi[s][j] = hi;
        blo[s][j] = (__bf16)r;
      }
    }

    #pragma unroll
    for (int m = 0; m < 4; ++m) {
      v16bf ahi = *(const v16bf*)(&sA[m * 2 + 0][lane][0]);
      v16bf alo = *(const v16bf*)(&sA[m * 2 + 1][lane][0]);
      #pragma unroll
      for (int s = 0; s < 2; ++s) {
        acc[m][s] = __builtin_amdgcn_wmma_f32_16x16x32_bf16(
            false, ahi, false, bhi[s], (short)0, acc[m][s], false, false);
        acc[m][s] = __builtin_amdgcn_wmma_f32_16x16x32_bf16(
            false, alo, false, bhi[s], (short)0, acc[m][s], false, false);
        acc[m][s] = __builtin_amdgcn_wmma_f32_16x16x32_bf16(
            false, ahi, false, blo[s], (short)0, acc[m][s], false, false);
      }
    }
  }

  // C layout: VGPR r -> row r (lanes<16) / row r+8 (lanes>=16), col = lane&15
  #pragma unroll
  for (int m = 0; m < 4; ++m)
    #pragma unroll
    for (int s = 0; s < 2; ++s) {
      int col = v0 + s * 16 + nlo;
      #pragma unroll
      for (int r = 0; r < 8; ++r) {
        int row = m * 16 + r + 8 * hh;
        logits[(size_t)row * VV + col] = acc[m][s][r];
      }
    }
}

// ---------------------------------------------------------------------------
// Per (row, 2000-col chunk): max, sum(exp(x-max)), exact top-4.
// ---------------------------------------------------------------------------
__global__ void __launch_bounds__(256)
k_partial(const float* __restrict__ logits, RowChunk* __restrict__ part) {
  int c = blockIdx.x;   // 0..15
  int r = blockIdx.y;   // 0..63
  int t = threadIdx.x;
  const float* rowp = logits + (size_t)r * VV + c * 2000;

  float vals[8];
  float lmax = FLOW;
  float tv[4]; int ti[4];
  #pragma unroll
  for (int q = 0; q < 4; ++q) { tv[q] = FLOW; ti[q] = 0x7fffffff; }

  #pragma unroll
  for (int it = 0; it < 8; ++it) {
    int pos = it * 256 + t;
    float x = (pos < 2000) ? rowp[pos] : FLOW;
    vals[it] = x;
    lmax = fmaxf(lmax, x);
    if (pos < 2000) ins4(tv, ti, x, c * 2000 + pos);
  }

  __shared__ float sred[256];
  sred[t] = lmax;
  __syncthreads();
  for (int st = 128; st > 0; st >>= 1) {
    if (t < st) sred[t] = fmaxf(sred[t], sred[t + st]);
    __syncthreads();
  }
  float gmax = sred[0];
  __syncthreads();

  float ls = 0.f;
  #pragma unroll
  for (int it = 0; it < 8; ++it) {
    int pos = it * 256 + t;
    if (pos < 2000) ls += __expf(vals[it] - gmax);
  }
  sred[t] = ls;
  __syncthreads();
  for (int st = 128; st > 0; st >>= 1) {
    if (t < st) sred[t] += sred[t + st];
    __syncthreads();
  }
  float gsum = sred[0];

  __shared__ float s4v[256][4];
  __shared__ int   s4i[256][4];
  #pragma unroll
  for (int q = 0; q < 4; ++q) { s4v[t][q] = tv[q]; s4i[t][q] = ti[q]; }
  __syncthreads();
  for (int st = 128; st > 0; st >>= 1) {
    if (t < st) merge4(&s4v[t][0], &s4i[t][0], &s4v[t + st][0], &s4i[t + st][0]);
    __syncthreads();
  }
  if (t == 0) {
    RowChunk rc;
    rc.mx = gmax; rc.sm = gsum;
    #pragma unroll
    for (int q = 0; q < 4; ++q) { rc.v[q] = s4v[0][q]; rc.ix[q] = s4i[0][q]; }
    part[r * 16 + c] = rc;
  }
}

// ---------------------------------------------------------------------------
// Final per-step select: LSE + row top4 -> beam combine -> token reorder ->
// build next step's A fragments.
// ---------------------------------------------------------------------------
__global__ void __launch_bounds__(256)
k_select(const RowChunk* __restrict__ part,
         float* scores, int* tokens, int* last,
         const float* __restrict__ emb, const float* __restrict__ pool,
         __bf16* Ahi, __bf16* Alo, int step) {
  __shared__ float rlse[MM];
  __shared__ float rtv[MM][4];
  __shared__ int   rti[MM][4];
  __shared__ int   newtok[NB][BEAM][TSTEPS];
  __shared__ float nsc[MM];
  __shared__ int   nlast[MM];
  int t = threadIdx.x;

  if (t < MM) {
    float gm = FLOW;
    for (int c = 0; c < 16; ++c) gm = fmaxf(gm, part[t * 16 + c].mx);
    float s = 0.f;
    float mv[4]; int mi[4];
    #pragma unroll
    for (int q = 0; q < 4; ++q) { mv[q] = FLOW; mi[q] = 0x7fffffff; }
    for (int c = 0; c < 16; ++c) {
      RowChunk rc = part[t * 16 + c];
      s += rc.sm * __expf(rc.mx - gm);
      merge4(mv, mi, rc.v, rc.ix);
    }
    rlse[t] = gm + __logf(s);
    #pragma unroll
    for (int q = 0; q < 4; ++q) { rtv[t][q] = mv[q]; rti[t][q] = mi[q]; }
  }
  __syncthreads();

  if (t < NB) {
    int b = t;
    float cv[4]; int cc[4];
    #pragma unroll
    for (int q = 0; q < 4; ++q) { cv[q] = FLOW; cc[q] = 0x7fffffff; }
    for (int k = 0; k < BEAM; ++k) {
      int row = b * BEAM + k;
      float off = scores[row] - rlse[row];
      #pragma unroll
      for (int j = 0; j < 4; ++j)
        ins4(cv, cc, off + rtv[row][j], k * VV + rti[row][j]);
    }
    for (int nk = 0; nk < BEAM; ++nk) {
      int beam = cc[nk] / VV;
      int word = cc[nk] - beam * VV;
      for (int p = 0; p < TSTEPS; ++p)
        newtok[b][nk][p] = tokens[(b * BEAM + beam) * TSTEPS + p];
      newtok[b][nk][step] = word;
      nsc[b * BEAM + nk]   = cv[nk];
      nlast[b * BEAM + nk] = word;
    }
  }
  __syncthreads();

  for (int i = t; i < NB * BEAM * TSTEPS; i += 256)
    tokens[i] = ((const int*)newtok)[i];
  if (t < MM) { scores[t] = nsc[t]; last[t] = nlast[t]; }
  __syncthreads();

  buildA(emb, pool, last, Ahi, Alo, t);
}

__global__ void k_output(const int* __restrict__ tokens,
                         const float* __restrict__ scores,
                         float* __restrict__ out) {
  int t = threadIdx.x + blockIdx.x * blockDim.x;
  if (t < 512) {
    int b = t >> 5, p = t & 31;
    ((int*)out)[t] = tokens[(b * BEAM + 0) * TSTEPS + p];   // top beam tokens
  } else if (t < 576) {
    out[t] = scores[t - 512];                               // beam scores
  }
}

// ---------------------------------------------------------------------------
extern "C" void kernel_launch(void* const* d_in, const int* in_sizes, int n_in,
                              void* d_out, int out_size, void* d_ws, size_t ws_size,
                              hipStream_t stream) {
  const float* enc   = (const float*)d_in[0];   // [16,128,1024]
  const float* mask  = (const float*)d_in[1];   // [16,128]
  const int*   tgt   = (const int*)  d_in[2];   // [16]
  const float* emb   = (const float*)d_in[3];   // [32000,1024]
  const float* W     = (const float*)d_in[4];   // [1024,32000]
  // d_in[5] = max_steps (scalar, fixed at 32)

  char* ws = (char*)d_ws;
  float*    pool   = (float*)   (ws + OFF_POOL);
  __bf16*   Ahi    = (__bf16*)  (ws + OFF_AHI);
  __bf16*   Alo    = (__bf16*)  (ws + OFF_ALO);
  float*    logits = (float*)   (ws + OFF_LOGITS);
  RowChunk* part   = (RowChunk*)(ws + OFF_PART);
  float*    scores = (float*)   (ws + OFF_SCORES);
  int*      tokens = (int*)     (ws + OFF_TOKENS);
  int*      last   = (int*)     (ws + OFF_LAST);

  k_encpool<<<NB, 256, 0, stream>>>(enc, mask, pool);
  k_init<<<1, 256, 0, stream>>>(tgt, emb, pool, tokens, scores, last, Ahi, Alo);

  for (int i = 1; i < TSTEPS; ++i) {
    k_gemm<<<VV / 256, 256, 0, stream>>>(W, Ahi, Alo, logits);
    dim3 g(16, MM);
    k_partial<<<g, 256, 0, stream>>>(logits, part);
    k_select<<<1, 256, 0, stream>>>(part, scores, tokens, last, emb, pool,
                                    Ahi, Alo, i);
  }
  k_output<<<3, 256, 0, stream>>>(tokens, scores, (float*)d_out);
}
